// MultiHeadAttention_54202487276107
// MI455X (gfx1250) — matrix operand order repaired and checked by reading, not compiled, not run on hardware
//
#include <hip/hip_runtime.h>
#include <hip/hip_bf16.h>
#include <math.h>

typedef _Float16 v16h __attribute__((ext_vector_type(16)));
typedef _Float16 v8h  __attribute__((ext_vector_type(8)));
typedef float    v8f  __attribute__((ext_vector_type(8)));
typedef unsigned int u32x4 __attribute__((ext_vector_type(4)));
typedef unsigned int u32x8 __attribute__((ext_vector_type(8)));

#define NH 16
#define KVH 4
#define HD 128
#define BB 2
#define SS 2048
#define EE 2048
#define QKVO 3072

#define SHUF16(lo, hi) __builtin_shufflevector((lo), (hi), 0,1,2,3,4,5,6,7,8,9,10,11,12,13,14,15)

static __device__ __forceinline__ v8f wmma_f16(v16h a, v16h b, v8f c) {
  return __builtin_amdgcn_wmma_f32_16x16x32_f16(false, a, false, b, (short)0, c, false, false);
}

// LDS byte offset of a __shared__ object (generic -> addrspace(3) -> int).
static __device__ __forceinline__ unsigned lds_addr_u32(const void* p) {
  return (unsigned)(unsigned long long)(__attribute__((address_space(3))) const void*)p;
}

// Tensor Data Mover: async 2-D f16 tile load global -> LDS (ISA ch.7/8).
// Loads `rows` rows of 128 halves (row stride `row_stride` halves in memory),
// padding LDS rows to 144 halves (pad_interval=64 DW, pad_amount=8 DW).
static __device__ __forceinline__ void tdm_load_k_tile(const _Float16* gptr,
                                                       unsigned lds_byte,
                                                       unsigned rows,
                                                       unsigned row_stride,
                                                       unsigned tensor_rows) {
  unsigned long long g = (unsigned long long)gptr;
  u32x4 g0;
  g0[0] = 1u;                                   // count=1 (valid), no gather
  g0[1] = lds_byte;                             // lds_addr
  g0[2] = (unsigned)g;                          // global_addr[31:0]
  g0[3] = (unsigned)((g >> 32) & 0x1FFFFFFu)    // global_addr[56:32]
        | (2u << 30);                           // type = 2 ("image")
  u32x8 g1;
  g1[0] = (1u << 16)                            // data_size = 2 bytes
        | (1u << 20)                            // pad_enable
        | (5u << 22)                            // pad_interval: 64 DWORDs (one row)
        | (7u << 25);                           // pad_amount: 8 DWORDs (16 halves)
  g1[1] = (128u & 0xFFFFu) << 16;               // tensor_dim0[15:0] = 128
  g1[2] = (tensor_rows & 0xFFFFu) << 16;        // tensor_dim0 hi=0 | tensor_dim1 lo
  g1[3] = ((tensor_rows >> 16) & 0xFFFFu)       // tensor_dim1 hi
        | (128u << 16);                         // tile_dim0 = 128
  g1[4] = rows;                                 // tile_dim1 (tile_dim2 = 0 -> 2D)
  g1[5] = row_stride;                           // tensor_dim0_stride[31:0]
  g1[6] = 0u;
  g1[7] = 0u;
  asm volatile("tensor_load_to_lds %0, %1" :: "s"(g0), "s"(g1) : "memory");
}

// ---------------------------------------------------------------- convert
__global__ void f32_to_f16_kernel(const float* __restrict__ in,
                                  _Float16* __restrict__ out, long n) {
  long i = (long)blockIdx.x * blockDim.x + threadIdx.x;
  long stride = (long)gridDim.x * blockDim.x;
  for (; i < n; i += stride) out[i] = (_Float16)in[i];
}

// ---------------------------------------------------------------- WMMA GEMM
// C[M,N] = A[M,K] * B[K,N], f16 in, f32 accumulate. M mult of 128, N of 64, K of 32.
// Block = 128 threads (4 waves). Block tile 128x64, K-step 32.
// Each wave computes a 32x64 slab: 2 A-fragments reuse 4 B-fragments -> 8 WMMA/step.
template <bool STORE_HALF>
__global__ __launch_bounds__(128, 1)
void wmma_gemm_kernel(const _Float16* __restrict__ A, const _Float16* __restrict__ Bm,
                      void* __restrict__ Cv, int M, int N, int K) {
  __shared__ _Float16 a_lds[128][48];   // [m][k], 96B row stride (32B aligned)
  __shared__ _Float16 bt_lds[64][48];   // [n][k] (B transposed)
  const int tid  = threadIdx.x;
  const int w    = tid >> 5;
  const int lane = tid & 31;
  const int hlf  = lane >> 4;
  const int n16  = lane & 15;
  const int m0 = blockIdx.y * 128;
  const int n0 = blockIdx.x * 64;

  v8f acc[2][4];
#pragma unroll
  for (int s2 = 0; s2 < 2; ++s2)
#pragma unroll
    for (int nt = 0; nt < 4; ++nt) acc[s2][nt] = v8f{};

  for (int k0 = 0; k0 < K; k0 += 32) {
    // cooperative A load: 128x32 halves, 512 chunks of 8 halves
#pragma unroll
    for (int i = 0; i < 4; ++i) {
      int c   = i * 128 + tid;         // 0..511
      int row = c >> 2;                // 0..127
      int col = (c & 3) * 8;           // 0..24
      v8h va = *(const v8h*)(A + (long)(m0 + row) * K + k0 + col);
      *(v8h*)(&a_lds[row][col]) = va;
    }
    // cooperative B load (transposed into LDS): 32x64 halves
#pragma unroll
    for (int i = 0; i < 2; ++i) {
      int c    = i * 128 + tid;        // 0..255
      int krow = c >> 3;               // 0..31
      int col  = (c & 7) * 8;          // 0..56
      v8h vb = *(const v8h*)(Bm + (long)(k0 + krow) * N + n0 + col);
#pragma unroll
      for (int j = 0; j < 8; ++j) bt_lds[col + j][krow] = vb[j];
    }
    __syncthreads();

    // A fragments: e[i]=K(hlf*16+i)
    v16h af[2];
#pragma unroll
    for (int s2 = 0; s2 < 2; ++s2) {
      int arow = w * 32 + s2 * 16 + n16;
      v8h alo = *(const v8h*)(&a_lds[arow][hlf * 16]);
      v8h ahi = *(const v8h*)(&a_lds[arow][hlf * 16 + 8]);
      af[s2] = SHUF16(alo, ahi);
    }
#pragma unroll
    for (int nt = 0; nt < 4; ++nt) {
      int bcol = nt * 16 + n16;
      v16h bfrag = *(const v16h*)(&bt_lds[bcol][hlf * 16]); // e[i]=B[hlf*16+i][col]
#pragma unroll
      for (int s2 = 0; s2 < 2; ++s2)
        acc[s2][nt] = wmma_f16(af[s2], bfrag, acc[s2][nt]);
    }
    __syncthreads();
  }

  // C layout: VGPR r -> M = r + 8*hlf ; N = n16
#pragma unroll
  for (int s2 = 0; s2 < 2; ++s2) {
#pragma unroll
    for (int nt = 0; nt < 4; ++nt) {
#pragma unroll
      for (int r = 0; r < 8; ++r) {
        int row = m0 + w * 32 + s2 * 16 + r + hlf * 8;
        int col = n0 + nt * 16 + n16;
        if constexpr (STORE_HALF)
          ((_Float16*)Cv)[(long)row * N + col] = (_Float16)acc[s2][nt][r];
        else
          ((float*)Cv)[(long)row * N + col] = acc[s2][nt][r];
      }
    }
  }
}

// ---------------------------------------------------------------- RoPE + GQA pack
__global__ void rope_pack_kernel(const _Float16* __restrict__ qkv,
                                 _Float16* __restrict__ qh,
                                 _Float16* __restrict__ kh,
                                 _Float16* __restrict__ vh) {
  long idx = (long)blockIdx.x * blockDim.x + threadIdx.x;
  long total = (long)BB * SS * 24 * 64;
  if (idx >= total) return;
  int d  = (int)(idx & 63);
  long t = idx >> 6;
  int hh = (int)(t % 24);
  long row = t / 24;
  int s = (int)(row % SS);
  int b = (int)(row / SS);
  const _Float16* src = qkv + row * QKVO + hh * HD;
  if (hh < 20) {
    float x1 = (float)src[d];
    float x2 = (float)src[d + 64];
    float invfreq = __powf(10000.0f, -(float)d / 64.0f);  // theta^(-2d/128)
    float ang = (float)s * invfreq;
    float sn, cs;
    __sincosf(ang, &sn, &cs);
    float o1 = x1 * cs - x2 * sn;
    float o2 = x2 * cs + x1 * sn;
    _Float16* dst;
    if (hh < 16) dst = qh + (((long)(b * NH + hh) * SS) + s) * HD;
    else         dst = kh + (((long)(b * KVH + (hh - 16)) * SS) + s) * HD;
    dst[d]      = (_Float16)o1;
    dst[d + 64] = (_Float16)o2;
  } else {
    _Float16* dst = vh + (((long)(b * KVH + (hh - 20)) * SS) + s) * HD;
    dst[d]      = src[d];
    dst[d + 64] = src[d + 64];
  }
}

// ---------------------------------------------------------------- flash attention
// Block = 128 threads (4 waves), handles (b, h, 64 query rows).
// K tile staged via the Tensor Data Mover (async, TENSORcnt); V staged transposed
// by VALU; scores + P*V on v_wmma_f32_16x16x32_f16 with online softmax.
__global__ __launch_bounds__(128, 1)
void flash_attn_kernel(const _Float16* __restrict__ qh,
                       const _Float16* __restrict__ kh,
                       const _Float16* __restrict__ vh,
                       _Float16* __restrict__ attn) {
  __shared__ _Float16 ks[32][144];      // K tile [key][d], 288B stride (TDM-padded)
  __shared__ _Float16 vt[128][48];      // V tile transposed [d][key]
  __shared__ _Float16 pst[4][16][48];   // per-wave P staging [m][k]
  const int tid  = threadIdx.x;
  const int w    = tid >> 5;
  const int lane = tid & 31;
  const int hlf  = lane >> 4;
  const int n16  = lane & 15;
  const int qb = blockIdx.x;            // 0..S/64-1
  const int h  = blockIdx.y;
  const int b  = blockIdx.z;
  const int kvh = h / (NH / KVH);
  const long qbase  = (long)(b * NH + h) * SS * HD;
  const long kvbase = (long)(b * KVH + kvh) * SS * HD;
  const int q0 = qb * 64 + w * 16;
  const float scale = 0.08838834764831845f;  // 1/sqrt(128)
  const unsigned lds_ks = lds_addr_u32(&ks[0][0]);

  // Q A-fragments (4 chunks of K=32 over head dim), kept in registers
  v16h qa[4];
  {
    const _Float16* qrow = qh + qbase + (long)(q0 + n16) * HD;
#pragma unroll
    for (int c = 0; c < 4; ++c) {
      v8h lo = *(const v8h*)(qrow + c * 32 + hlf * 16);
      v8h hi = *(const v8h*)(qrow + c * 32 + hlf * 16 + 8);
      qa[c] = SHUF16(lo, hi);
    }
  }

  v8f o[8];
#pragma unroll
  for (int dt = 0; dt < 8; ++dt) o[dt] = v8f{};
  float mrow[8], lsum[8];
#pragma unroll
  for (int r = 0; r < 8; ++r) { mrow[r] = -1e30f; lsum[r] = 0.0f; }

  const int kb_end = qb * 2 + 2;  // cover keys [0, qb*64+64)
  for (int kb = 0; kb < kb_end; ++kb) {
    // K tile: async TDM copy (one descriptor, issued by wave 0 only)
    if (w == 0) {
      tdm_load_k_tile(kh + kvbase + (long)(kb * 32) * HD, lds_ks, 32u, (unsigned)HD,
                      (unsigned)SS);
      __builtin_amdgcn_s_wait_tensorcnt(0);
    }
    // V tile: cooperative load + transpose into LDS
#pragma unroll
    for (int i = 0; i < 4; ++i) {
      int c   = i * 128 + tid;    // 0..511
      int row = c >> 4;           // 0..31
      int col = (c & 15) * 8;     // 0..120
      v8h vv = *(const v8h*)(vh + kvbase + (long)(kb * 32 + row) * HD + col);
#pragma unroll
      for (int j = 0; j < 8; ++j) vt[col + j][row] = vv[j];
    }
    if (kb + 1 < kb_end)
      __builtin_prefetch(vh + kvbase + (long)(kb * 32 + 32) * HD + tid * 8, 0, 0);
    __syncthreads();

    if (kb * 32 <= q0 + 15) {  // wave-uniform: block has unmasked keys for this wave
      // scores: two 16x16 tiles over 32 keys
      v8f st[2] = {v8f{}, v8f{}};
#pragma unroll
      for (int sub = 0; sub < 2; ++sub) {
#pragma unroll
        for (int c = 0; c < 4; ++c) {
          // B-frag: N = key (sub*16+n16), e[i] = K[key][c*32 + hlf*16 + i]
          v16h bfrag = *(const v16h*)(&ks[sub * 16 + n16][c * 32 + hlf * 16]);
          st[sub] = wmma_f16(qa[c], bfrag, st[sub]);
        }
      }
      // scale + causal mask + online softmax; stage P for the P*V GEMM
#pragma unroll
      for (int r = 0; r < 8; ++r) {
        int m = r + hlf * 8;
        int qrow = q0 + m;
        float s0 = st[0][r] * scale;
        float s1 = st[1][r] * scale;
        if (kb * 32 + n16 > qrow)      s0 = -1e30f;
        if (kb * 32 + 16 + n16 > qrow) s1 = -1e30f;
        float rmax = fmaxf(s0, s1);
#pragma unroll
        for (int msk = 1; msk < 16; msk <<= 1)
          rmax = fmaxf(rmax, __shfl_xor(rmax, msk, 32));
        float newmax = fmaxf(mrow[r], rmax);
        float alpha = __expf(mrow[r] - newmax);
        float p0 = __expf(s0 - newmax);
        float p1 = __expf(s1 - newmax);
        float rsum = p0 + p1;
#pragma unroll
        for (int msk = 1; msk < 16; msk <<= 1)
          rsum += __shfl_xor(rsum, msk, 32);
        lsum[r] = lsum[r] * alpha + rsum;
        mrow[r] = newmax;
#pragma unroll
        for (int dt = 0; dt < 8; ++dt) o[dt][r] *= alpha;
        pst[w][m][n16]      = (_Float16)p0;
        pst[w][m][16 + n16] = (_Float16)p1;
      }
      // P A-fragment (16x32): row = n16, K = 32 key positions
      v8h plo = *(const v8h*)(&pst[w][n16][hlf * 16]);
      v8h phi = *(const v8h*)(&pst[w][n16][hlf * 16 + 8]);
      v16h pfrag = SHUF16(plo, phi);
#pragma unroll
      for (int dt = 0; dt < 8; ++dt) {
        // B-frag: N = d col (dt*16+n16), e[i] = V[hlf*16+i][dcol] = vt[dcol][hlf*16+i]
        v16h vfrag = *(const v16h*)(&vt[dt * 16 + n16][hlf * 16]);
        o[dt] = wmma_f16(pfrag, vfrag, o[dt]);
      }
    }
    __syncthreads();
  }

  // normalize + store attn[B*S, NH*HD] f16
#pragma unroll
  for (int r = 0; r < 8; ++r) {
    int s = q0 + r + hlf * 8;
    float inv = 1.0f / lsum[r];
    long orow = (long)(b * SS + s) * (NH * HD) + h * HD;
#pragma unroll
    for (int dt = 0; dt < 8; ++dt)
      attn[orow + dt * 16 + n16] = (_Float16)(o[dt][r] * inv);
  }
}

// ---------------------------------------------------------------- launch
extern "C" void kernel_launch(void* const* d_in, const int* in_sizes, int n_in,
                              void* d_out, int out_size, void* d_ws, size_t ws_size,
                              hipStream_t stream) {
  (void)in_sizes; (void)n_in; (void)out_size; (void)ws_size;
  const float* x       = (const float*)d_in[0];
  const float* w_qkv   = (const float*)d_in[1];
  const float* w_dense = (const float*)d_in[2];

  char* ws = (char*)d_ws;
  size_t off = 0;
  auto alloc_h = [&](size_t halves) {
    _Float16* p = (_Float16*)(ws + off);
    off += ((halves * sizeof(_Float16) + 255) / 256) * 256;
    return p;
  };
  _Float16* x_h      = alloc_h((size_t)BB * SS * EE);
  _Float16* wqkv_h   = alloc_h((size_t)EE * QKVO);
  _Float16* wdense_h = alloc_h((size_t)(NH * HD) * EE);
  _Float16* qkv_h    = alloc_h((size_t)BB * SS * QKVO);
  _Float16* q_h      = alloc_h((size_t)BB * NH * SS * HD);
  _Float16* k_h      = alloc_h((size_t)BB * KVH * SS * HD);
  _Float16* v_h      = alloc_h((size_t)BB * KVH * SS * HD);
  _Float16* attn_h   = alloc_h((size_t)BB * SS * NH * HD);

  // 1) f32 -> f16 (halves GEMM traffic, enables the v_wmma f16 path)
  f32_to_f16_kernel<<<2048, 256, 0, stream>>>(x, x_h, (long)BB * SS * EE);
  f32_to_f16_kernel<<<2048, 256, 0, stream>>>(w_qkv, wqkv_h, (long)EE * QKVO);
  f32_to_f16_kernel<<<2048, 256, 0, stream>>>(w_dense, wdense_h, (long)(NH * HD) * EE);

  // 2) fused QKV projection: [4096,2048] x [2048,3072] -> f16
  wmma_gemm_kernel<true><<<dim3(QKVO / 64, (BB * SS) / 128), 128, 0, stream>>>(
      x_h, wqkv_h, qkv_h, BB * SS, QKVO, EE);

  // 3) RoPE + GQA head packing
  long nrope = (long)BB * SS * 24 * 64;
  rope_pack_kernel<<<(unsigned)((nrope + 255) / 256), 256, 0, stream>>>(
      qkv_h, q_h, k_h, v_h);

  // 4) causal flash attention (TDM-staged K tiles, never materializes scores)
  flash_attn_kernel<<<dim3(SS / 64, NH, BB), 128, 0, stream>>>(q_h, k_h, v_h, attn_h);

  // 5) output projection: [4096,2048] x [2048,2048] -> f32 out
  wmma_gemm_kernel<false><<<dim3(EE / 64, (BB * SS) / 128), 128, 0, stream>>>(
      attn_h, wdense_h, d_out, BB * SS, EE, NH * HD);
}
